// Decoder_Rasha_73830487818685
// MI455X (gfx1250) — matrix operand order repaired and checked by reading, not compiled, not run on hardware
//
#include <hip/hip_runtime.h>
#include <hip/hip_bf16.h>

// ---------------- problem constants (from reference) ----------------
#define BB   128
#define LL   196
#define ENC  512
#define HH   512
#define EE   256
#define VV   10000
#define TT   20
#define KCAT 1280          // E + ENC + H
#define N4H  2048          // 4*H

typedef __bf16          v16bf __attribute__((ext_vector_type(16)));
typedef float           v8f   __attribute__((ext_vector_type(8)));
typedef unsigned short  us8   __attribute__((ext_vector_type(8)));

union Frag { v16bf v; us8 h[2]; };

// ---------------- helpers ----------------
__device__ __forceinline__ unsigned short f2bf(float f) {
    unsigned int u = __float_as_uint(f);
    unsigned int r = u + 0x7FFFu + ((u >> 16) & 1u);   // round-to-nearest-even
    return (unsigned short)(r >> 16);
}
__device__ __forceinline__ float bf2f(unsigned short h) {
    return __uint_as_float(((unsigned int)h) << 16);
}
__device__ __forceinline__ float fast_tanh(float x) {
#if __has_builtin(__builtin_amdgcn_tanhf)
    return __builtin_amdgcn_tanhf(x);      // native v_tanh_f32 (CDNA5 TRANS op)
#else
    return tanhf(x);
#endif
}
__device__ __forceinline__ float fast_rcp(float x) {
#if __has_builtin(__builtin_amdgcn_rcpf)
    return __builtin_amdgcn_rcpf(x);       // native v_rcp_f32
#else
    return 1.0f / x;
#endif
}
__device__ __forceinline__ float sigf(float x) { return fast_rcp(1.0f + __expf(-x)); }

// ---------------- generic bf16 WMMA GEMM ----------------
// C[M,N](f32 and/or bf16) = act( A[M,K]_bf16 * Bt[N,K]_bf16^T + bias[N] )
// grid = (ceil(N/64), M/32), block = 256 (8 waves, each one 16x16 tile)
#define ACT_NONE 0
#define ACT_TANH 1
#define ACT_SIG  2
__global__ void gemm_bf16_wmma(const unsigned short* __restrict__ A,
                               const unsigned short* __restrict__ Bt,
                               const float* __restrict__ bias,
                               float* __restrict__ C,
                               unsigned short* __restrict__ Cbf,
                               int M, int N, int K,
                               int lda, int ldb, long ldc, int act)
{
    const int lane = threadIdx.x & 31;
    const int wave = threadIdx.x >> 5;
    const int tile_m = blockIdx.y * 32 + (wave >> 2) * 16;
    const int tile_n = blockIdx.x * 64 + (wave & 3) * 16;
    if (tile_m >= M || tile_n >= N) return;            // wave-uniform: EXEC stays all-ones

    // A fragment addressing (16x32 bf16): row = lane&15, kbase = (lane>=16)*8
    const int arow = tile_m + (lane & 15);
    const int akb  = (lane >> 4) << 3;
    // B fragment from transposed weight: col = lane&15, kbase = (lane>=16)*8
    const int bcol = tile_n + (lane & 15);
    const int bkb  = (lane >> 4) << 3;

    const unsigned short* pa = A  + (size_t)arow * lda + akb;
    const unsigned short* pb = Bt + (size_t)bcol * ldb + bkb;

    v8f acc = {};
    for (int k = 0; k < K; k += 32) {
        Frag a, b;
        a.h[0] = *(const us8*)(pa + k);        // K = kb+0..7
        a.h[1] = *(const us8*)(pa + k + 16);   // K = 16+kb+0..7
        b.h[0] = *(const us8*)(pb + k);        // K = kb+0..7
        b.h[1] = *(const us8*)(pb + k + 16);   // K = 16+kb+0..7
        acc = __builtin_amdgcn_wmma_f32_16x16x32_bf16(
            false, a.v, false, b.v, (short)0, acc, false, false);
    }

    const int n     = tile_n + (lane & 15);
    const int mbase = tile_m + ((lane >> 4) << 3);
    const float bv  = bias ? bias[n] : 0.0f;
#pragma unroll
    for (int r = 0; r < 8; ++r) {
        float v = acc[r] + bv;
        if (act == ACT_TANH)     v = fast_tanh(v);
        else if (act == ACT_SIG) v = sigf(v);
        long idx = (long)(mbase + r) * ldc + n;
        if (C)   C[idx]   = v;
        if (Cbf) Cbf[idx] = f2bf(v);
    }
}

// ---------------- prep kernels ----------------
__global__ void f32_to_bf16_kernel(const float* __restrict__ src,
                                   unsigned short* __restrict__ dst, long n) {
    long i = (long)blockIdx.x * 256 + threadIdx.x;
    if (i < n) dst[i] = f2bf(src[i]);
}

// W[K,N] f32  ->  Wt[N,K] bf16
__global__ void transpose_bf16_kernel(const float* __restrict__ W,
                                      unsigned short* __restrict__ Wt, int K, int N) {
    long idx = (long)blockIdx.x * 256 + threadIdx.x;
    if (idx >= (long)K * N) return;
    int n = (int)(idx / K);
    int k = (int)(idx - (long)n * K);
    Wt[idx] = f2bf(W[(long)k * N + n]);
}

// Wcat_t[N4H][KCAT]: k<768 -> W_ih[k][n], else W_hh[k-768][n]
__global__ void build_wcat_kernel(const float* __restrict__ Wih,
                                  const float* __restrict__ Whh,
                                  unsigned short* __restrict__ Wt) {
    long idx = (long)blockIdx.x * 256 + threadIdx.x;
    if (idx >= (long)N4H * KCAT) return;
    int n = (int)(idx / KCAT);
    int k = (int)(idx - (long)n * KCAT);
    float v = (k < EE + ENC) ? Wih[(long)k * N4H + n] : Whh[(long)(k - (EE + ENC)) * N4H + n];
    Wt[idx] = f2bf(v);
}

__global__ void bias_cat_kernel(const float* __restrict__ bih,
                                const float* __restrict__ bhh,
                                float* __restrict__ bc) {
    int n = blockIdx.x * 256 + threadIdx.x;
    if (n < N4H) bc[n] = bih[n] + bhh[n];
}

__global__ void feat_mean_kernel(const float* __restrict__ img,
                                 float* __restrict__ fm,
                                 unsigned short* __restrict__ fm_bf) {
    int b = blockIdx.x;
    for (int e = threadIdx.x; e < ENC; e += 256) {
        float s = 0.f;
        for (int l = 0; l < LL; ++l) s += img[((size_t)b * LL + l) * ENC + e];
        float v = s * (1.0f / (float)LL);
        fm[b * ENC + e]    = v;
        fm_bf[b * ENC + e] = f2bf(v);
    }
}

// ---------------- attention: scores + softmax + context ----------------
__global__ void attention_kernel(const unsigned short* __restrict__ enc_proj, // [B,L,H] bf16 (b_ae added)
                                 const float* __restrict__ dec_proj,          // [B,H]
                                 const float* __restrict__ v_att,
                                 const float* __restrict__ b_v,
                                 const float* __restrict__ img_feat,          // [B,L,ENC] f32
                                 float* __restrict__ alphas,                  // [B,T,L]
                                 float* __restrict__ context,                 // [B,ENC]
                                 int t)
{
    const int b    = blockIdx.x;
    const int tid  = threadIdx.x;
    const int lane = tid & 31;
    const int wave = tid >> 5;
    __shared__ float es[LL];
    __shared__ float red[256];
    __shared__ float s_dp[HH];   // dec_proj row, reused 196x
    __shared__ float s_va[HH];   // v_att, reused 196x

    for (int k = tid; k < HH; k += 256) {
        s_dp[k] = dec_proj[(size_t)b * HH + k];
        s_va[k] = v_att[k];
    }
    __syncthreads();

    for (int l = wave; l < LL; l += 8) {
        const unsigned short* ep = enc_proj + ((size_t)b * LL + l) * HH;
        float s = 0.f;
        for (int k = lane; k < HH; k += 32)
            s += fast_tanh(bf2f(ep[k]) + s_dp[k]) * s_va[k];
        for (int off = 16; off > 0; off >>= 1) s += __shfl_xor(s, off, 32);
        if (lane == 0) es[l] = s + b_v[0];
    }
    __syncthreads();

    float m = -1e30f;
    for (int l = tid; l < LL; l += 256) m = fmaxf(m, es[l]);
    red[tid] = m; __syncthreads();
    for (int s = 128; s > 0; s >>= 1) { if (tid < s) red[tid] = fmaxf(red[tid], red[tid + s]); __syncthreads(); }
    m = red[0]; __syncthreads();

    float sum = 0.f;
    for (int l = tid; l < LL; l += 256) { float e = __expf(es[l] - m); es[l] = e; sum += e; }
    red[tid] = sum; __syncthreads();
    for (int s = 128; s > 0; s >>= 1) { if (tid < s) red[tid] += red[tid + s]; __syncthreads(); }
    float inv = fast_rcp(red[0]);
    __syncthreads();

    for (int l = tid; l < LL; l += 256) {
        float a = es[l] * inv;
        es[l] = a;
        alphas[((size_t)b * TT + t) * LL + l] = a;
    }
    __syncthreads();

    const float* imf = img_feat + (size_t)b * LL * ENC;
    for (int e0 = tid; e0 < ENC; e0 += 256) {
        float s = 0.f;
        for (int l = 0; l < LL; ++l) s += es[l] * imf[(size_t)l * ENC + e0];
        context[(size_t)b * ENC + e0] = s;
    }
}

// ---------------- x = concat(emb[cap], ctx*gate, h) in bf16 ----------------
__global__ void build_x_kernel(const float* __restrict__ emb,
                               const int* __restrict__ captions, int t,
                               const float* __restrict__ context,
                               const float* __restrict__ gate,
                               const float* __restrict__ h,
                               unsigned short* __restrict__ x)   // [B,KCAT]
{
    int b = blockIdx.x;
    int cap = captions[b * TT + t];
    for (int j = threadIdx.x; j < KCAT; j += 256) {
        float v;
        if (j < EE)            v = emb[(size_t)cap * EE + j];
        else if (j < EE + ENC) v = context[(size_t)b * ENC + (j - EE)] * gate[(size_t)b * ENC + (j - EE)];
        else                   v = h[(size_t)b * HH + (j - EE - ENC)];
        x[(size_t)b * KCAT + j] = f2bf(v);
    }
}

// ---------------- LSTM pointwise update ----------------
__global__ void lstm_update_kernel(const float* __restrict__ gates, // [B,4H] (i,f,g,o)
                                   float* __restrict__ c,
                                   float* __restrict__ h,
                                   unsigned short* __restrict__ h_bf)
{
    int idx = blockIdx.x * 256 + threadIdx.x;   // B*H
    int b = idx >> 9, j = idx & 511;
    const float* g = gates + (size_t)b * N4H;
    float ig = g[j], fg = g[HH + j], gg = g[2 * HH + j], og = g[3 * HH + j];
    float cn = sigf(fg) * c[idx] + sigf(ig) * fast_tanh(gg);
    float hn = sigf(og) * fast_tanh(cn);
    c[idx] = cn; h[idx] = hn; h_bf[idx] = f2bf(hn);
}

// ---------------- host side ----------------
static inline size_t align_up(size_t x, size_t a) { return (x + a - 1) & ~(a - 1); }

extern "C" void kernel_launch(void* const* d_in, const int* in_sizes, int n_in,
                              void* d_out, int out_size, void* d_ws, size_t ws_size,
                              hipStream_t stream) {
    (void)in_sizes; (void)n_in; (void)out_size; (void)ws_size;
    const float* img_feat = (const float*)d_in[0];
    const float* W_h   = (const float*)d_in[1];  const float* b_h   = (const float*)d_in[2];
    const float* W_c   = (const float*)d_in[3];  const float* b_c   = (const float*)d_in[4];
    const float* W_gate= (const float*)d_in[5];  const float* b_gate= (const float*)d_in[6];
    const float* emb   = (const float*)d_in[7];
    const float* W_ae  = (const float*)d_in[8];  const float* b_ae  = (const float*)d_in[9];
    const float* W_ad  = (const float*)d_in[10]; const float* b_ad  = (const float*)d_in[11];
    const float* v_att = (const float*)d_in[12]; const float* b_v   = (const float*)d_in[13];
    const float* W_ih  = (const float*)d_in[14]; const float* b_ih  = (const float*)d_in[15];
    const float* W_hh  = (const float*)d_in[16]; const float* b_hh  = (const float*)d_in[17];
    const float* W_out = (const float*)d_in[18]; const float* b_out = (const float*)d_in[19];
    const int*   captions = (const int*)d_in[20];

    float* outs   = (float*)d_out;                          // [B,T,V]
    float* alphas = outs + (size_t)BB * TT * VV;            // [B,T,L]

    // workspace carve-up
    char* p = (char*)d_ws; size_t off = 0;
    auto alloc = [&](size_t bytes) { size_t o = off; off = align_up(off + bytes, 256); return (void*)(p + o); };
    unsigned short* img_bf   = (unsigned short*)alloc((size_t)BB * LL * ENC * 2);
    unsigned short* enc_bf   = (unsigned short*)alloc((size_t)BB * LL * HH * 2);
    unsigned short* Wh_t     = (unsigned short*)alloc((size_t)HH * ENC * 2);
    unsigned short* Wc_t     = (unsigned short*)alloc((size_t)HH * ENC * 2);
    unsigned short* Wae_t    = (unsigned short*)alloc((size_t)HH * ENC * 2);
    unsigned short* Wad_t    = (unsigned short*)alloc((size_t)HH * HH * 2);
    unsigned short* Wgate_t  = (unsigned short*)alloc((size_t)ENC * HH * 2);
    unsigned short* Wcat_t   = (unsigned short*)alloc((size_t)N4H * KCAT * 2);
    unsigned short* Wout_t   = (unsigned short*)alloc((size_t)VV * HH * 2);
    float*          bias_cat = (float*)alloc((size_t)N4H * 4);
    float*          fm       = (float*)alloc((size_t)BB * ENC * 4);
    unsigned short* fm_bf    = (unsigned short*)alloc((size_t)BB * ENC * 2);
    float*          h_f      = (float*)alloc((size_t)BB * HH * 4);
    float*          c_f      = (float*)alloc((size_t)BB * HH * 4);
    unsigned short* h_bf     = (unsigned short*)alloc((size_t)BB * HH * 2);
    float*          decp     = (float*)alloc((size_t)BB * HH * 4);
    float*          context  = (float*)alloc((size_t)BB * ENC * 4);
    float*          gate     = (float*)alloc((size_t)BB * ENC * 4);
    unsigned short* x_bf     = (unsigned short*)alloc((size_t)BB * KCAT * 2);
    float*          gates    = (float*)alloc((size_t)BB * N4H * 4);

    dim3 blk(256);

    // ---- prep ----
    {
        long n = (long)BB * LL * ENC;
        f32_to_bf16_kernel<<<dim3((unsigned)((n + 255) / 256)), blk, 0, stream>>>(img_feat, img_bf, n);
    }
    transpose_bf16_kernel<<<dim3((ENC * HH + 255) / 256), blk, 0, stream>>>(W_h,   Wh_t,   ENC, HH);
    transpose_bf16_kernel<<<dim3((ENC * HH + 255) / 256), blk, 0, stream>>>(W_c,   Wc_t,   ENC, HH);
    transpose_bf16_kernel<<<dim3((ENC * HH + 255) / 256), blk, 0, stream>>>(W_ae,  Wae_t,  ENC, HH);
    transpose_bf16_kernel<<<dim3((HH * HH  + 255) / 256), blk, 0, stream>>>(W_ad,  Wad_t,  HH,  HH);
    transpose_bf16_kernel<<<dim3((HH * ENC + 255) / 256), blk, 0, stream>>>(W_gate,Wgate_t,HH,  ENC);
    transpose_bf16_kernel<<<dim3((unsigned)(((long)HH * VV + 255) / 256)), blk, 0, stream>>>(W_out, Wout_t, HH, VV);
    build_wcat_kernel<<<dim3((unsigned)(((long)N4H * KCAT + 255) / 256)), blk, 0, stream>>>(W_ih, W_hh, Wcat_t);
    bias_cat_kernel<<<dim3((N4H + 255) / 256), blk, 0, stream>>>(b_ih, b_hh, bias_cat);
    feat_mean_kernel<<<dim3(BB), blk, 0, stream>>>(img_feat, fm, fm_bf);

    // ---- h0 / c0 ----
    gemm_bf16_wmma<<<dim3(HH / 64, BB / 32), blk, 0, stream>>>(
        fm_bf, Wh_t, b_h, h_f, h_bf, BB, HH, ENC, ENC, ENC, HH, ACT_TANH);
    gemm_bf16_wmma<<<dim3(HH / 64, BB / 32), blk, 0, stream>>>(
        fm_bf, Wc_t, b_c, c_f, nullptr, BB, HH, ENC, ENC, ENC, HH, ACT_TANH);

    // ---- enc_proj = img_feat @ W_ae + b_ae  (bf16 result) ----
    gemm_bf16_wmma<<<dim3(HH / 64, (BB * LL) / 32), blk, 0, stream>>>(
        img_bf, Wae_t, b_ae, nullptr, enc_bf, BB * LL, HH, ENC, ENC, ENC, HH, ACT_NONE);

    // ---- scan over T ----
    for (int t = 0; t < TT; ++t) {
        // dec_proj = h @ W_ad + b_ad
        gemm_bf16_wmma<<<dim3(HH / 64, BB / 32), blk, 0, stream>>>(
            h_bf, Wad_t, b_ad, decp, nullptr, BB, HH, HH, HH, HH, HH, ACT_NONE);
        // attention + softmax + context
        attention_kernel<<<dim3(BB), blk, 0, stream>>>(
            enc_bf, decp, v_att, b_v, img_feat, alphas, context, t);
        // gate = sigmoid(h @ W_gate + b_gate)
        gemm_bf16_wmma<<<dim3(ENC / 64, BB / 32), blk, 0, stream>>>(
            h_bf, Wgate_t, b_gate, gate, nullptr, BB, ENC, HH, HH, HH, ENC, ACT_SIG);
        // x = [emb[cap], ctx*gate, h]
        build_x_kernel<<<dim3(BB), blk, 0, stream>>>(emb, captions, t, context, gate, h_f, x_bf);
        // gates = x @ Wcat + (b_ih + b_hh)
        gemm_bf16_wmma<<<dim3(N4H / 64, BB / 32), blk, 0, stream>>>(
            x_bf, Wcat_t, bias_cat, gates, nullptr, BB, N4H, KCAT, KCAT, KCAT, N4H, ACT_NONE);
        // LSTM pointwise
        lstm_update_kernel<<<dim3((BB * HH) / 256), blk, 0, stream>>>(gates, c_f, h_f, h_bf);
        // out[:, t, :] = h_new @ W_out + b_out  (strided into d_out)
        gemm_bf16_wmma<<<dim3((VV + 63) / 64, BB / 32), blk, 0, stream>>>(
            h_bf, Wout_t, b_out, outs + (size_t)t * VV, nullptr,
            BB, VV, HH, HH, HH, (long)TT * VV, ACT_NONE);
    }
}
